// GNNModel_68822555951662
// MI455X (gfx1250) — matrix operand order repaired and checked by reading, not compiled, not run on hardware
//
#include <hip/hip_runtime.h>
#include <hip/hip_bf16.h>
#include <math.h>

typedef __attribute__((ext_vector_type(16))) __bf16 v16bf;
typedef __attribute__((ext_vector_type(8)))  float  v8f;

constexpr int IN_DIM = 256, HID = 128, NHEAD = 4, CHAN = 32, NLAYER = 2, NOUT = 40;
constexpr float NEG_SLOPE = 0.2f;
constexpr int KCH  = 64;    // K-chunk staged per LDS buffer
constexpr int KSTR = 72;    // padded K stride in packed weights / LDS (16B aligned, bank spread)

// ---- order-preserving float <-> uint encode for atomicMax-based segment max ----
__device__ __forceinline__ unsigned fenc(float f) {
    unsigned u = __float_as_uint(f);
    return (u & 0x80000000u) ? ~u : (u | 0x80000000u);
}
__device__ __forceinline__ float fdec(unsigned e) {
    unsigned u = (e & 0x80000000u) ? (e & 0x7FFFFFFFu) : ~e;
    return __uint_as_float(u);
}

// =====================================================================
// Weight pre-pack: B[Kc,Nc] f32 row-major -> bf16, fragment-native,
// chunk-contiguous col-major-with-pad:
//   Wt[((chunk*Nc)+col)*KSTR + k] = bf16(B[(chunk*KCH+k)*Nc + col])
// Each 64-K chunk is then a single contiguous block -> async b128 memcpy to LDS.
// =====================================================================
__global__ void pack_weights(const float* __restrict__ B, __bf16* __restrict__ Wt,
                             int Kc, int Nc)
{
    int idx = blockIdx.x * blockDim.x + threadIdx.x;
    if (idx >= Kc * Nc) return;
    int c   = idx / (Nc * KCH);
    int r   = idx - c * (Nc * KCH);
    int col = r / KCH;
    int k   = r - col * KCH;
    Wt[((long long)c * Nc + col) * KSTR + k] = (__bf16)B[(c * KCH + k) * Nc + col];
}

// ---- async global->LDS copy of one packed weight chunk (16B packets) ----
__device__ __forceinline__ void stage_chunk_async(const __bf16* __restrict__ Bt, int chunk,
                                                  unsigned ldsBase, int nBytes, int tid)
{
    const unsigned gbase = (unsigned)chunk * (unsigned)nBytes;   // packed weights < 128KB
    for (int i = tid * 16; i < nBytes; i += 256 * 16) {
        unsigned lds  = ldsBase + (unsigned)i;
        unsigned voff = gbase + (unsigned)i;
        asm volatile("global_load_async_to_lds_b128 %0, %1, %2"
                     :: "v"(lds), "v"(voff), "s"(Bt) : "memory");
    }
}

// =====================================================================
// WMMA strip GEMM: C[M,Nc] = A[M,Kc] @ B[Kc,Nc] (+ bias), bf16 WMMA
// 16x16x32, fp32 accumulate. One wave per 16-row strip holding all Nc/16
// accumulator tiles. Packed bf16 weights are streamed to double-buffered
// LDS with GLOBAL_LOAD_ASYNC_TO_LDS_B128 (copy of chunk c+1 overlaps the
// WMMAs of chunk c; s_wait_asynccnt + barrier close each stage). Each
// lane's B fragment is two contiguous ds_load_b128's; A fragments are two
// global_load_b128 (bf16 A) or b128+cvt (f32 A). Optional bf16 mirror of C.
// =====================================================================
template<int Kc, int Nc, typename AT>
__global__ void __launch_bounds__(256)
wmma_gemm_strip(const AT* __restrict__ A, const __bf16* __restrict__ Bt,
                const float* __restrict__ bias, float* __restrict__ C,
                __bf16* __restrict__ Cb, int M)
{
    constexpr int TILES_N     = (Nc + 15) / 16;
    constexpr int NCHUNK      = Kc / KCH;
    constexpr int CHUNK_BYTES = Nc * KSTR * 2;
    __shared__ __align__(16) __bf16 Bs[2][Nc * KSTR];

    const int tid   = threadIdx.x;
    const int wid   = tid >> 5;
    const int lane  = tid & 31;
    const int l16   = lane & 15;
    const int half  = lane >> 4;
    const int strip = blockIdx.x * (blockDim.x >> 5) + wid;
    const int row0  = strip * 16;
    const bool active = (row0 < M);            // wave-uniform: EXEC full at WMMA

    v8f acc[TILES_N] = {};

    const int       kb   = half * 16;          // A-fragment K sub-offset (contiguous K order, same as B)
    const long long arow = row0 + l16;
    // low 32 bits of a generic LDS pointer == LDS byte offset (ISA 10.2 aperture rule)
    const unsigned ldsBase0 = (unsigned)(uintptr_t)&Bs[0][0];

    stage_chunk_async(Bt, 0, ldsBase0, CHUNK_BYTES, tid);
    asm volatile("s_wait_asynccnt 0x0" ::: "memory");
    __syncthreads();

    for (int c = 0; c < NCHUNK; ++c) {
        if (c + 1 < NCHUNK)                    // overlap next chunk's copy with WMMAs
            stage_chunk_async(Bt, c + 1, ldsBase0 + ((c + 1) & 1) * CHUNK_BYTES,
                              CHUNK_BYTES, tid);
        if (active) {
            const __bf16* Bcur = &Bs[c & 1][0];
            const int kc = c * KCH;
            if (c + 1 < NCHUNK)                // gfx1250 global_prefetch of next A chunk
                __builtin_prefetch(A + arow * Kc + (kc + KCH + kb), 0, 3);
            #pragma unroll
            for (int k0 = 0; k0 < KCH; k0 += 32) {
                // A fragment, contiguous K order matching B (lanes 0-15: K 0-15;
                // lanes 16-31: K 16-31)
                v16bf a;
                const AT* Ap = A + arow * Kc + (kc + k0 + kb);
                if constexpr (sizeof(AT) == 2) {               // bf16 A: 2x b128 loads
                    ((uint4*)&a)[0] = *(const uint4*)(Ap);
                    ((uint4*)&a)[1] = *(const uint4*)(Ap + 8);
                } else {                                       // f32 A: load + convert
                    #pragma unroll
                    for (int j = 0; j < 8; ++j) {
                        a[j]     = (__bf16)Ap[j];
                        a[8 + j] = (__bf16)Ap[8 + j];
                    }
                }
                #pragma unroll
                for (int t = 0; t < TILES_N; ++t) {
                    const int col = t * 16 + l16;
                    // B fragment: lanes 0-15 hold K 0-15, lanes 16-31 hold K 16-31
                    // of column `col` -> 16 contiguous bf16 in LDS (2x ds_load_b128)
                    v16bf b;
                    if (col < Nc) {
                        const uint4* bp =
                            (const uint4*)(Bcur + col * KSTR + k0 + half * 16);
                        ((uint4*)&b)[0] = bp[0];
                        ((uint4*)&b)[1] = bp[1];
                    } else {
                        #pragma unroll
                        for (int m = 0; m < 16; ++m) b[m] = (__bf16)0.f;
                    }
                    acc[t] = __builtin_amdgcn_wmma_f32_16x16x32_bf16(
                                 false, a, false, b, (short)0, acc[t], false, false);
                }
            }
        }
        asm volatile("s_wait_asynccnt 0x0" ::: "memory");
        __syncthreads();
    }
    if (!active) return;
    // C/D layout: VGPR i, lanes 0-15 -> (M=i, N=lane); lanes 16-31 -> (M=8+i, N=lane-16)
    #pragma unroll
    for (int t = 0; t < TILES_N; ++t) {
        const int col = t * 16 + l16;
        if (col >= Nc) continue;
        const float bv = bias ? bias[col] : 0.f;
        const int r0 = row0 + half * 8;
        #pragma unroll
        for (int i = 0; i < 8; ++i) {
            const float v = acc[t][i] + bv;
            const long long idx = (long long)(r0 + i) * Nc + col;
            C[idx] = v;
            if (Cb) Cb[idx] = (__bf16)v;
        }
    }
}

// ---------------------------------------------------------------------
__global__ void fill_zero(float* __restrict__ p, long long n) {
    long long i = (long long)blockIdx.x * blockDim.x + threadIdx.x;
    if (i < n) p[i] = 0.f;
}

// per-(node,head) attention logits: al_s = <h, att_src>, al_d = <h, att_dst>
__global__ void attn_scores(const float* __restrict__ hl,
                            const float* __restrict__ as_, const float* __restrict__ ad_,
                            float* __restrict__ als, float* __restrict__ ald, int N)
{
    int idx = blockIdx.x * blockDim.x + threadIdx.x;
    if (idx >= N * NHEAD) return;
    int n = idx >> 2, hd = idx & 3;
    const float* hp = hl + (long long)n * HID + hd * CHAN;
    float s = 0.f, d = 0.f;
    #pragma unroll
    for (int c = 0; c < CHAN; ++c) {
        float v = hp[c];
        s += v * as_[hd * CHAN + c];
        d += v * ad_[hd * CHAN + c];
    }
    als[idx] = s; ald[idx] = d;
}

__device__ __forceinline__ void edge_sd(int e, int E0,
                                        const int* __restrict__ src,
                                        const int* __restrict__ dst, int& s, int& d) {
    if (e < E0) { s = src[e]; d = dst[e]; } else { s = d = e - E0; }
}

__global__ void edge_max_kernel(const int* __restrict__ src, const int* __restrict__ dst,
                                int E0, int Etot,
                                const float* __restrict__ als, const float* __restrict__ ald,
                                unsigned* __restrict__ menc)
{
    int idx = blockIdx.x * blockDim.x + threadIdx.x;
    if (idx >= Etot * NHEAD) return;
    int e = idx >> 2, hd = idx & 3, s, d;
    edge_sd(e, E0, src, dst, s, d);
    float v = als[s * NHEAD + hd] + ald[d * NHEAD + hd];
    v = (v > 0.f) ? v : NEG_SLOPE * v;
    atomicMax(&menc[d * NHEAD + hd], fenc(v));
}

__global__ void edge_sum_kernel(const int* __restrict__ src, const int* __restrict__ dst,
                                int E0, int Etot,
                                const float* __restrict__ als, const float* __restrict__ ald,
                                const unsigned* __restrict__ menc, float* __restrict__ denom)
{
    int idx = blockIdx.x * blockDim.x + threadIdx.x;
    if (idx >= Etot * NHEAD) return;
    int e = idx >> 2, hd = idx & 3, s, d;
    edge_sd(e, E0, src, dst, s, d);
    float v = als[s * NHEAD + hd] + ald[d * NHEAD + hd];
    v = (v > 0.f) ? v : NEG_SLOPE * v;
    float m = fdec(menc[d * NHEAD + hd]);
    atomicAdd(&denom[d * NHEAD + hd], expf(v - m));
}

// one wave per edge: lane l handles channels 4l..4l+3 (head = l>>3); float4 gather + 4 atomics
__global__ void edge_aggregate(const int* __restrict__ src, const int* __restrict__ dst,
                               int E0, int Etot,
                               const float* __restrict__ hl,
                               const float* __restrict__ als, const float* __restrict__ ald,
                               const unsigned* __restrict__ menc,
                               const float* __restrict__ denom, float* __restrict__ agg)
{
    long long gtid = (long long)blockIdx.x * blockDim.x + threadIdx.x;
    int e = (int)(gtid >> 5);
    if (e >= Etot) return;
    int lane = threadIdx.x & 31, s, d;
    edge_sd(e, E0, src, dst, s, d);
    int hd = lane >> 3;
    float v = als[s * NHEAD + hd] + ald[d * NHEAD + hd];
    v = (v > 0.f) ? v : NEG_SLOPE * v;
    float m = fdec(menc[d * NHEAD + hd]);
    float alpha = expf(v - m) / (denom[d * NHEAD + hd] + 1e-16f);
    const float4 hv = *(const float4*)(hl + (long long)s * HID + lane * 4);
    float* op = agg + (long long)d * HID + lane * 4;
    atomicAdd(op + 0, hv.x * alpha);
    atomicAdd(op + 1, hv.y * alpha);
    atomicAdd(op + 2, hv.z * alpha);
    atomicAdd(op + 3, hv.w * alpha);
}

// h += relu(agg + bias); also refresh the bf16 mirror used as WMMA A-operand
__global__ void residual_relu(float* __restrict__ h, __bf16* __restrict__ hb,
                              const float* __restrict__ agg,
                              const float* __restrict__ bias, int total)
{
    int i = blockIdx.x * blockDim.x + threadIdx.x;
    if (i >= total) return;
    float v = agg[i] + bias[i & (HID - 1)];
    float nv = h[i] + ((v > 0.f) ? v : 0.f);
    h[i]  = nv;
    hb[i] = (__bf16)nv;
}

// in-place log_softmax over rows of 40, one wave32 per row
__global__ void log_softmax_rows(float* __restrict__ logits, int rows, int cols)
{
    int wid  = (blockIdx.x * blockDim.x + threadIdx.x) >> 5;
    int lane = threadIdx.x & 31;
    if (wid >= rows) return;
    float* row = logits + (long long)wid * cols;
    float v0 = (lane < cols)      ? row[lane]      : -3.4e38f;
    float v1 = (lane + 32 < cols) ? row[lane + 32] : -3.4e38f;
    float mx = fmaxf(v0, v1);
    #pragma unroll
    for (int off = 16; off; off >>= 1) mx = fmaxf(mx, __shfl_xor(mx, off, 32));
    float s = ((lane < cols) ? expf(v0 - mx) : 0.f) +
              ((lane + 32 < cols) ? expf(v1 - mx) : 0.f);
    #pragma unroll
    for (int off = 16; off; off >>= 1) s += __shfl_xor(s, off, 32);
    float lse = logf(s);
    if (lane < cols)      row[lane]      = v0 - mx - lse;
    if (lane + 32 < cols) row[lane + 32] = v1 - mx - lse;
}

// =====================================================================
extern "C" void kernel_launch(void* const* d_in, const int* in_sizes, int n_in,
                              void* d_out, int out_size, void* d_ws, size_t ws_size,
                              hipStream_t stream)
{
    const float* x     = (const float*)d_in[0];
    const int*   eidx  = (const int*)d_in[1];
    const float* W_in  = (const float*)d_in[2];
    const float* b_in  = (const float*)d_in[3];
    const float* W_gat = (const float*)d_in[4];
    const float* att_s = (const float*)d_in[5];
    const float* att_d = (const float*)d_in[6];
    const float* b_gat = (const float*)d_in[7];
    const float* W_out = (const float*)d_in[8];
    const float* b_out = (const float*)d_in[9];

    const int N    = in_sizes[0] / IN_DIM;
    const int E0   = in_sizes[1] / 2;
    const int Etot = E0 + N;               // edges + self loops
    const int* src = eidx;                 // edge_index[0]
    const int* dstp = eidx + E0;           // edge_index[1]

    // workspace layout (agg, menc, denom contiguous so one zero-fill covers them)
    float*    h     = (float*)d_ws;
    float*    hl    = h   + (long long)N * HID;
    float*    agg   = hl  + (long long)N * HID;
    unsigned* menc  = (unsigned*)(agg + (long long)N * HID);
    float*    denom = (float*)(menc + (long long)N * NHEAD);
    float*    als   = denom + (long long)N * NHEAD;
    float*    ald   = als   + (long long)N * NHEAD;
    __bf16*   hb    = (__bf16*)(ald + (long long)N * NHEAD);   // bf16 mirror of h
    // packed bf16 weights (chunk-contiguous, padded col-major)
    __bf16*   wt_in  = hb + (long long)N * HID;
    __bf16*   wt_gat = wt_in  + (IN_DIM / KCH) * HID * KSTR;
    __bf16*   wt_out = wt_gat + (long long)NLAYER * (HID / KCH) * HID * KSTR;
    float*    logits = (float*)d_out;

    // ---- pre-pack all weights into WMMA-fragment-native bf16 layout ----
    pack_weights<<<(IN_DIM * HID + 255) / 256, 256, 0, stream>>>(W_in, wt_in, IN_DIM, HID);
    for (int l = 0; l < NLAYER; ++l)
        pack_weights<<<(HID * HID + 255) / 256, 256, 0, stream>>>(
            W_gat + (long long)l * HID * HID,
            wt_gat + (long long)l * (HID / KCH) * HID * KSTR, HID, HID);
    pack_weights<<<(HID * NOUT + 255) / 256, 256, 0, stream>>>(W_out, wt_out, HID, NOUT);

    const int strips = (N + 15) / 16;
    const int gemmBlocks = (strips + 7) / 8;

    // h = x @ W_in + b_in   (also produces bf16 mirror hb)
    wmma_gemm_strip<IN_DIM, HID, float><<<gemmBlocks, 256, 0, stream>>>(
        x, wt_in, b_in, h, hb, N);

    for (int l = 0; l < NLAYER; ++l) {
        wmma_gemm_strip<HID, HID, __bf16><<<gemmBlocks, 256, 0, stream>>>(
            hb, wt_gat + (long long)l * (HID / KCH) * HID * KSTR, nullptr, hl, nullptr, N);

        long long zn = (long long)N * HID + 2LL * N * NHEAD;   // agg + menc + denom
        fill_zero<<<(int)((zn + 255) / 256), 256, 0, stream>>>(agg, zn);

        attn_scores<<<(N * NHEAD + 255) / 256, 256, 0, stream>>>(
            hl, att_s + l * NHEAD * CHAN, att_d + l * NHEAD * CHAN, als, ald, N);

        int eth = (Etot * NHEAD + 255) / 256;
        edge_max_kernel<<<eth, 256, 0, stream>>>(src, dstp, E0, Etot, als, ald, menc);
        edge_sum_kernel<<<eth, 256, 0, stream>>>(src, dstp, E0, Etot, als, ald, menc, denom);

        long long aggThreads = (long long)Etot * 32;
        edge_aggregate<<<(int)((aggThreads + 255) / 256), 256, 0, stream>>>(
            src, dstp, E0, Etot, hl, als, ald, menc, denom, agg);

        residual_relu<<<(N * HID + 255) / 256, 256, 0, stream>>>(
            h, hb, agg, b_gat + l * HID, N * HID);
    }

    // logits = h @ W_out + b_out  (written straight into d_out), then log_softmax in place
    wmma_gemm_strip<HID, NOUT, __bf16><<<gemmBlocks, 256, 0, stream>>>(
        hb, wt_out, b_out, logits, nullptr, N);
    log_softmax_rows<<<(N + 7) / 8, 256, 0, stream>>>(logits, N, NOUT);
}
